// TestMoDLayer_50689204027432
// MI455X (gfx1250) — matrix operand order repaired and checked by reading, not compiled, not run on hardware
//
#include <hip/hip_runtime.h>
#include <hip/hip_bf16.h>

// ---------------------------------------------------------------------------
// MoD layer for MI455X (gfx1250, wave32, WMMA f32_16x16x32_f16,
// async-to-LDS staging for attention K/V tiles)
// ---------------------------------------------------------------------------

typedef __attribute__((ext_vector_type(16))) _Float16 v16h;
typedef __attribute__((ext_vector_type(8)))  _Float16 v8h;
typedef __attribute__((ext_vector_type(4)))  _Float16 v4h;
typedef __attribute__((ext_vector_type(8)))  float    v8f;

#define B_    4
#define NTOK  4096
#define DIM   1024
#define CAP   2048          // N * 0.5
#define H_    16
#define DH    64            // DIM / H_

union F16V { v16h v; v8h h[2]; };

__device__ __forceinline__ v8f v8f_zero() {
  v8f z = {0.f,0.f,0.f,0.f,0.f,0.f,0.f,0.f};
  return z;
}

__device__ __forceinline__ v8f wmma_f16(v16h a, v16h b, v8f c) {
  // D = A(16x32 f16) * B(32x16 f16) + C(16x16 f32)
  return __builtin_amdgcn_wmma_f32_16x16x32_f16(false, a, false, b, (short)0, c,
                                                false, false);
}

// A fragment (16x32 f16). lane: row = lane&15, h8 = lane>>4.
// elements 0..7  -> K = h8*16 + (0..7)     (contiguous 16B)
// elements 8..15 -> K = h8*16 + 8 + (0..7) (contiguous 16B)
__device__ __forceinline__ v16h load_a_frag(const _Float16* rowbase, int kbase, int h8) {
  F16V f;
  f.h[0] = *(const v8h*)(rowbase + kbase + h8 * 16);
  f.h[1] = *(const v8h*)(rowbase + kbase + h8 * 16 + 8);
  return f.v;
}

// Async copy 16 bytes global -> LDS (tracked by ASYNCcnt, no VGPR round-trip).
__device__ __forceinline__ void async_b128(unsigned lds_off, const _Float16* gptr) {
  asm volatile("global_load_async_to_lds_b128 %0, %1, off"
               :: "v"(lds_off), "v"((unsigned long long)(uintptr_t)gptr)
               : "memory");
}

// ---------------------------------------------------------------------------
// Zero-fill the output (scatter target)
// ---------------------------------------------------------------------------
__global__ void mod_zero_kernel(float* __restrict__ p, size_t n) {
  size_t i = (size_t)blockIdx.x * blockDim.x + threadIdx.x;
  size_t stride = (size_t)gridDim.x * blockDim.x;
  for (; i < n; i += stride) p[i] = 0.0f;
}

// ---------------------------------------------------------------------------
// Router: gate = sigmoid(x @ Wg); exact top-CAP per batch via rank counting
// (matches jax.lax.top_k tie-break: higher gate first, then smaller index)
// ---------------------------------------------------------------------------
__global__ __launch_bounds__(1024) void mod_router_kernel(
    const float* __restrict__ tok, const float* __restrict__ Wg,
    int* __restrict__ sel_idx, float* __restrict__ gate_sel) {
  __shared__ float g[NTOK];
  const int b    = blockIdx.x;
  const int tid  = threadIdx.x;
  const int lane = tid & 31;
  const int wave = tid >> 5;            // 32 waves of 32 lanes

  for (int t = wave; t < NTOK; t += 32) {
    const float* row = tok + ((size_t)b * NTOK + t) * DIM;
    float acc = 0.f;
    for (int k = lane; k < DIM; k += 32) acc += row[k] * Wg[k];
    acc += __shfl_xor(acc, 16, 32);
    acc += __shfl_xor(acc, 8, 32);
    acc += __shfl_xor(acc, 4, 32);
    acc += __shfl_xor(acc, 2, 32);
    acc += __shfl_xor(acc, 1, 32);
    if (lane == 0) g[t] = 1.0f / (1.0f + __expf(-acc));
  }
  __syncthreads();

  for (int n = tid; n < NTOK; n += 1024) {
    const float gv = g[n];
    int rank = 0;
    for (int m = 0; m < NTOK; ++m) {
      const float gm = g[m];
      rank += (gm > gv) || (gm == gv && m < n);
    }
    if (rank < CAP) {
      sel_idx[b * CAP + rank]  = n;
      gate_sel[b * CAP + rank] = gv;
    }
  }
}

// ---------------------------------------------------------------------------
// Gather routed tokens, convert f32 -> f16 (row-major [B*CAP, DIM])
// ---------------------------------------------------------------------------
__global__ __launch_bounds__(256) void mod_gather_kernel(
    const float* __restrict__ tok, const int* __restrict__ sel,
    _Float16* __restrict__ Xg) {
  const int bc = blockIdx.x;            // 0..B*CAP-1
  const int b  = bc >> 11;              // CAP = 2048
  const int n  = sel[bc];
  const float4* src = (const float4*)(tok + ((size_t)b * NTOK + n) * DIM);
  float4 v = src[threadIdx.x];
  v4h o;
  o[0] = (_Float16)v.x; o[1] = (_Float16)v.y;
  o[2] = (_Float16)v.z; o[3] = (_Float16)v.w;
  *(v4h*)(Xg + (size_t)bc * DIM + threadIdx.x * 4) = o;
}

// ---------------------------------------------------------------------------
// Weight transpose + f16 convert: Wt[n][k] = (f16) W[k][n]
// (B-matrix fragments then load as contiguous v16h)
// ---------------------------------------------------------------------------
__global__ __launch_bounds__(256) void mod_wtrans_kernel(
    const float* __restrict__ W, _Float16* __restrict__ Wt) {
  __shared__ __align__(16) _Float16 tile[64][65];
  const int k0 = blockIdx.x * 64;
  const int n0 = blockIdx.y * 64;
  const int tx = threadIdx.x & 63;
  const int ty = threadIdx.x >> 6;      // 0..3
  for (int i = ty; i < 64; i += 4)
    tile[i][tx] = (_Float16)W[(size_t)(k0 + i) * DIM + n0 + tx];
  __syncthreads();
  for (int i = ty; i < 64; i += 4)
    Wt[(size_t)(n0 + i) * DIM + k0 + tx] = tile[tx][i];
}

// ---------------------------------------------------------------------------
// WMMA GEMM: Y[M = B*CAP, N = DIM] = A @ W  (A f16 row-major, Wt pre-transposed)
// 8 waves/block, each wave a 32x64 tile. Modes select the epilogue:
//   0: Q / K -> f16 row-major [B*CAP, DIM]
//   2: V^T   -> f16 [B, H, DH, CAP]   (contiguous v8h stores along tokens)
//   3: O     -> f32 scatter out[b, sel_idx[c], :] = y * gate_sel (final output)
// ---------------------------------------------------------------------------
__global__ __launch_bounds__(256) void mod_gemm_kernel(
    const _Float16* __restrict__ A, const _Float16* __restrict__ Bt, int mode,
    _Float16* __restrict__ out16, float* __restrict__ out32,
    const int* __restrict__ sel_idx, const float* __restrict__ gate_sel) {
  const int wave = threadIdx.x >> 5;
  const int lane = threadIdx.x & 31;
  const int n15  = lane & 15;
  const int h8   = lane >> 4;
  const int wm   = wave & 1;
  const int wn   = wave >> 1;
  const int mbase = blockIdx.x * 64 + wm * 32;
  const int nbase = blockIdx.y * 256 + wn * 64;

  v8f acc[2][4];
#pragma unroll
  for (int mi = 0; mi < 2; ++mi)
#pragma unroll
    for (int j = 0; j < 4; ++j) acc[mi][j] = v8f_zero();

  const _Float16* arow0 = A + (size_t)(mbase + n15) * DIM;
  const _Float16* arow1 = arow0 + (size_t)16 * DIM;
  const _Float16* bcol0 = Bt + (size_t)(nbase + n15) * DIM;

  for (int kb = 0; kb < DIM; kb += 32) {
    const v16h a0 = load_a_frag(arow0, kb, h8);
    const v16h a1 = load_a_frag(arow1, kb, h8);
#pragma unroll
    for (int j = 0; j < 4; ++j) {
      const v16h bf = *(const v16h*)(bcol0 + (size_t)(j * 16) * DIM + kb + h8 * 16);
      acc[0][j] = wmma_f16(a0, bf, acc[0][j]);
      acc[1][j] = wmma_f16(a1, bf, acc[1][j]);
    }
  }

  // Epilogue. D fragment: lane n = n15, element r -> row M = base + r + h8*8.
#pragma unroll
  for (int mi = 0; mi < 2; ++mi) {
#pragma unroll
    for (int j = 0; j < 4; ++j) {
      const int Mb = mbase + mi * 16 + h8 * 8;      // 8-row group, never crosses batch
      const int n  = nbase + j * 16 + n15;
      if (mode == 0) {
#pragma unroll
        for (int r = 0; r < 8; ++r)
          out16[(size_t)(Mb + r) * DIM + n] = (_Float16)acc[mi][j][r];
      } else if (mode == 2) {
        const int h = n >> 6, d = n & 63;
        const int b = Mb >> 11, c0 = Mb & 2047;
        v8h pk;
#pragma unroll
        for (int r = 0; r < 8; ++r) pk[r] = (_Float16)acc[mi][j][r];
        *(v8h*)(out16 + (((size_t)b * H_ + h) * DH + d) * CAP + c0) = pk;
      } else {
        const int b = Mb >> 11, c0 = Mb & 2047;
#pragma unroll
        for (int r = 0; r < 8; ++r) {
          const int tokn = sel_idx[b * CAP + c0 + r];
          const float gs = gate_sel[b * CAP + c0 + r];
          out32[((size_t)b * NTOK + tokn) * DIM + n] = acc[mi][j][r] * gs;
        }
      }
    }
  }
}

// ---------------------------------------------------------------------------
// Flash attention over the routed tokens.
// grid = (CAP/128, B*H), 256 threads (8 waves); each wave owns 16 query rows.
// K/V tiles (32 keys) are staged in LDS with async-to-LDS b128 loads, double
// buffered (ASYNCcnt + barrier), shared by all 8 waves -> 8x less L2 traffic.
// S-fragment layout: lane = key(n15), element r = q row [+8 in upper half].
// P is transposed via per-wave LDS into an A fragment for the P @ V WMMAs.
// ---------------------------------------------------------------------------
__global__ __launch_bounds__(256) void mod_attn_kernel(
    const _Float16* __restrict__ Qb,   // [B*CAP, DIM] row-major
    const _Float16* __restrict__ Kb,   // [B*CAP, DIM] row-major (same as Q)
    const _Float16* __restrict__ Vt,   // [B, H, DH, CAP]
    _Float16* __restrict__ AO) {       // [B*CAP, DIM] row-major
  __shared__ __align__(16) _Float16 kts[2][32 * DH];   // [buf][key][dh]  4KB each
  __shared__ __align__(16) _Float16 vts[2][DH * 32];   // [buf][d][key]   4KB each
  __shared__ __align__(16) _Float16 pbuf[8][16 * 40];  // per-wave P tile (padded)
  const int tid  = threadIdx.x;
  const int wave = tid >> 5;
  const int lane = tid & 31;
  const int n15  = lane & 15;
  const int h8   = lane >> 4;
  const int b    = blockIdx.y >> 4;
  const int h    = blockIdx.y & 15;
  const int qrow = blockIdx.x * 128 + wave * 16;

  const _Float16* qptr = Qb + (size_t)(b * CAP + qrow + n15) * DIM + h * DH;
  const v16h aq0 = load_a_frag(qptr, 0, h8);    // K = 0..31 of dh
  const v16h aq1 = load_a_frag(qptr, 32, h8);   // K = 32..63 of dh
  const _Float16* Krow = Kb + (size_t)b * CAP * DIM + h * DH;   // row stride DIM
  const _Float16* Vh   = Vt + ((size_t)b * H_ + h) * DH * CAP;
  _Float16* pb = pbuf[wave];

  // Per-thread staging sources: one 16B chunk each for the K and V tiles.
  const _Float16* kgBase = Krow + (size_t)(tid >> 3) * DIM + (tid & 7) * 8;
  const _Float16* vgBase = Vh + (size_t)(tid >> 2) * CAP + (tid & 3) * 8;
  const unsigned kts_off = (unsigned)(uintptr_t)&kts[0][0] + tid * 16;
  const unsigned vts_off = (unsigned)(uintptr_t)&vts[0][0] + tid * 16;

  float m[8], l[8];
#pragma unroll
  for (int r = 0; r < 8; ++r) { m[r] = -1e30f; l[r] = 0.f; }
  v8f o[4];
#pragma unroll
  for (int j = 0; j < 4; ++j) o[j] = v8f_zero();

  const float sc = 0.18033688011112042f;  // (1/sqrt(64)) * log2(e)

  // Prologue: stage tile 0 into buffer 0.
  async_b128(kts_off, kgBase);
  async_b128(vts_off, vgBase);

  const int nIter = CAP / 32;
  for (int it = 0; it < nIter; ++it) {
    // Own async loads complete -> barrier makes everyone's tile visible.
    asm volatile("s_wait_asynccnt 0" ::: "memory");
    __syncthreads();

    // Stage next tile into the other buffer (overlaps with compute below).
    if (it + 1 < nIter) {
      const int nb = (it + 1) & 1;
      const int kbn = (it + 1) * 32;
      async_b128(kts_off + nb * 4096, kgBase + (size_t)kbn * DIM);
      async_b128(vts_off + nb * 4096, vgBase + kbn);
    }

    const int buf = it & 1;
    const int kb  = it * 32;
    const _Float16* Kt = kts[buf];
    const _Float16* Vb = vts[buf];

    // --- scores: two 16x16 tiles covering keys kb..kb+31 (B frags from LDS) ---
    const v16h bk00 = *(const v16h*)(Kt + n15 * DH + h8 * 16);
    const v16h bk01 = *(const v16h*)(Kt + n15 * DH + 32 + h8 * 16);
    const v16h bk10 = *(const v16h*)(Kt + (16 + n15) * DH + h8 * 16);
    const v16h bk11 = *(const v16h*)(Kt + (16 + n15) * DH + 32 + h8 * 16);
    v8f s0 = wmma_f16(aq0, bk00, v8f_zero());
    s0     = wmma_f16(aq1, bk01, s0);
    v8f s1 = wmma_f16(aq0, bk10, v8f_zero());
    s1     = wmma_f16(aq1, bk11, s1);

    // --- online softmax update (base-2 domain), write P to per-wave LDS ---
#pragma unroll
    for (int r = 0; r < 8; ++r) {
      const float t0 = s0[r] * sc;
      const float t1 = s1[r] * sc;
      float lm = fmaxf(t0, t1);
      lm = fmaxf(lm, __shfl_xor(lm, 1, 32));
      lm = fmaxf(lm, __shfl_xor(lm, 2, 32));
      lm = fmaxf(lm, __shfl_xor(lm, 4, 32));
      lm = fmaxf(lm, __shfl_xor(lm, 8, 32));
      const float mn   = fmaxf(m[r], lm);
      const float corr = exp2f(m[r] - mn);
      const float p0   = exp2f(t0 - mn);
      const float p1   = exp2f(t1 - mn);
      float rs = p0 + p1;
      rs += __shfl_xor(rs, 1, 32);
      rs += __shfl_xor(rs, 2, 32);
      rs += __shfl_xor(rs, 4, 32);
      rs += __shfl_xor(rs, 8, 32);
      l[r] = l[r] * corr + rs;
      m[r] = mn;
#pragma unroll
      for (int j = 0; j < 4; ++j) o[j][r] *= corr;
      const int row = r + h8 * 8;                 // q row of this element
      pb[row * 40 + n15]      = (_Float16)p0;     // key kb + n15
      pb[row * 40 + 16 + n15] = (_Float16)p1;     // key kb + 16 + n15
    }

    // LDS is in-order per wave; wait so the A-fragment reads see the stores.
    asm volatile("s_wait_dscnt 0" ::: "memory");

    F16V pa;
    pa.h[0] = *(const v8h*)(pb + n15 * 40 + h8 * 16);
    pa.h[1] = *(const v8h*)(pb + n15 * 40 + h8 * 16 + 8);

    // --- O += P(16x32) @ V(32x64), four 16-wide chunks of dh (B from LDS) ---
#pragma unroll
    for (int j = 0; j < 4; ++j) {
      const v16h bv = *(const v16h*)(Vb + (size_t)(j * 16 + n15) * 32 + h8 * 16);
      o[j] = wmma_f16(pa.v, bv, o[j]);
    }
  }

  // --- normalize and store (row-major f16, feature = h*64 + j*16 + n15) ---
#pragma unroll
  for (int r = 0; r < 8; ++r) {
    const float inv = 1.0f / l[r];
#pragma unroll
    for (int j = 0; j < 4; ++j) o[j][r] *= inv;
  }
#pragma unroll
  for (int j = 0; j < 4; ++j)
#pragma unroll
    for (int r = 0; r < 8; ++r)
      AO[(size_t)(b * CAP + qrow + r + h8 * 8) * DIM + h * DH + j * 16 + n15] =
          (_Float16)o[j][r];
}

// ---------------------------------------------------------------------------
// Host launch
// ---------------------------------------------------------------------------
extern "C" void kernel_launch(void* const* d_in, const int* in_sizes, int n_in,
                              void* d_out, int out_size, void* d_ws, size_t ws_size,
                              hipStream_t stream) {
  const float* tok = (const float*)d_in[0];
  const float* Wg  = (const float*)d_in[1];
  const float* Wq  = (const float*)d_in[2];
  const float* Wk  = (const float*)d_in[3];
  const float* Wv  = (const float*)d_in[4];
  const float* Wo  = (const float*)d_in[5];
  float* out = (float*)d_out;

  char* ws = (char*)d_ws;
  size_t off = 0;
  auto alloc = [&](size_t bytes) -> void* {
    off = (off + 255) & ~(size_t)255;
    void* p = ws + off;
    off += bytes;
    return p;
  };

  int*       sel_idx  = (int*)alloc((size_t)B_ * CAP * 4);
  float*     gate_sel = (float*)alloc((size_t)B_ * CAP * 4);
  _Float16*  Xg   = (_Float16*)alloc((size_t)B_ * CAP * DIM * 2);
  _Float16*  WqT  = (_Float16*)alloc((size_t)DIM * DIM * 2);
  _Float16*  WkT  = (_Float16*)alloc((size_t)DIM * DIM * 2);
  _Float16*  WvT  = (_Float16*)alloc((size_t)DIM * DIM * 2);
  _Float16*  WoT  = (_Float16*)alloc((size_t)DIM * DIM * 2);
  _Float16*  Qb   = (_Float16*)alloc((size_t)B_ * CAP * DIM * 2);
  _Float16*  Kb   = (_Float16*)alloc((size_t)B_ * CAP * DIM * 2);
  _Float16*  Vtb  = (_Float16*)alloc((size_t)B_ * CAP * DIM * 2);
  _Float16*  AO   = (_Float16*)alloc((size_t)B_ * CAP * DIM * 2);

  mod_zero_kernel<<<2048, 256, 0, stream>>>(out, (size_t)B_ * NTOK * DIM);
  mod_router_kernel<<<B_, 1024, 0, stream>>>(tok, Wg, sel_idx, gate_sel);
  mod_gather_kernel<<<B_ * CAP, 256, 0, stream>>>(tok, sel_idx, Xg);

  dim3 tg(DIM / 64, DIM / 64);
  mod_wtrans_kernel<<<tg, 256, 0, stream>>>(Wq, WqT);
  mod_wtrans_kernel<<<tg, 256, 0, stream>>>(Wk, WkT);
  mod_wtrans_kernel<<<tg, 256, 0, stream>>>(Wv, WvT);
  mod_wtrans_kernel<<<tg, 256, 0, stream>>>(Wo, WoT);

  dim3 gg((B_ * CAP) / 64, DIM / 256);
  mod_gemm_kernel<<<gg, 256, 0, stream>>>(Xg, WqT, 0, Qb, nullptr, nullptr, nullptr);
  mod_gemm_kernel<<<gg, 256, 0, stream>>>(Xg, WkT, 0, Kb, nullptr, nullptr, nullptr);
  mod_gemm_kernel<<<gg, 256, 0, stream>>>(Xg, WvT, 2, Vtb, nullptr, nullptr, nullptr);

  mod_attn_kernel<<<dim3(CAP / 128, B_ * H_), 256, 0, stream>>>(Qb, Kb, Vtb, AO);

  mod_gemm_kernel<<<gg, 256, 0, stream>>>(AO, WoT, 3, nullptr, out, sel_idx, gate_sel);
}
